// SimpleAttention_87694642250203
// MI455X (gfx1250) — matrix operand order repaired and checked by reading, not compiled, not run on hardware
//
#include <hip/hip_runtime.h>

#define Bd 4
#define Sd 2048
#define Ed 1024

typedef _Float16 v16h __attribute__((ext_vector_type(16)));
typedef _Float16 v8h  __attribute__((ext_vector_type(8)));
typedef _Float16 v4h  __attribute__((ext_vector_type(4)));
typedef float    v8f  __attribute__((ext_vector_type(8)));
typedef int      v4i  __attribute__((ext_vector_type(4)));
typedef unsigned int u32x4 __attribute__((ext_vector_type(4)));
typedef int      i32x8 __attribute__((ext_vector_type(8)));
typedef int      i32x4 __attribute__((ext_vector_type(4)));

#define LDK 40    // padded LDS row stride (halves) for 32-wide tiles (80B, 16B aligned)
#define LQH 520   // padded LDS half-row stride for Q (512 data + 8 pad halves)
#define QROW (2*LQH) // full Q row stride in halves (1040 -> 2080B, 16B aligned)
#define LKC 72    // padded LDS row stride for 64-wide K chunks (144B, 16B aligned)

// ---------------------------------------------------------------------------
// Feature probes
// ---------------------------------------------------------------------------
#if defined(__has_builtin)
#  if __has_builtin(__builtin_amdgcn_global_load_async_to_lds_b128)
#    define HAVE_ASYNC_LDS 1
#  endif
#  if __has_builtin(__builtin_amdgcn_tensor_load_to_lds)
#    define HAVE_TDM 1
#  endif
#endif
#ifndef HAVE_ASYNC_LDS
#  define HAVE_ASYNC_LDS 0
#endif
#ifndef HAVE_TDM
#  define HAVE_TDM 0
#endif

// ---------------------------------------------------------------------------
// Async global->LDS copy (GLOBAL_LOAD_ASYNC_TO_LDS_B128, ASYNCcnt) -- used as
// the staging fallback when the TDM builtin is unavailable.
// ---------------------------------------------------------------------------
__device__ __forceinline__ void async_ld_b128(void* lds, const void* g) {
#if HAVE_ASYNC_LDS
  __builtin_amdgcn_global_load_async_to_lds_b128((v4i*)g, (v4i*)lds, 0, 0);
#else
  *(v8h*)lds = *(const v8h*)g;
#endif
}

__device__ __forceinline__ void wait_async() {
#if HAVE_ASYNC_LDS
#  if __has_builtin(__builtin_amdgcn_s_wait_asynccnt)
  __builtin_amdgcn_s_wait_asynccnt(0);
#  else
  asm volatile("s_wait_asynccnt 0x0" ::: "memory");
#  endif
#endif
}

__device__ __forceinline__ void wait_tensor() {
#if HAVE_TDM
#  if __has_builtin(__builtin_amdgcn_s_wait_tensorcnt)
  __builtin_amdgcn_s_wait_tensorcnt(0);
#  else
  asm volatile("s_wait_tensorcnt 0x0" ::: "memory");
#  endif
#endif
}

// ---------------------------------------------------------------------------
// TDM: issue a 2D f16 tile load (TENSOR_LOAD_TO_LDS) described by a D#.
//  - tile_x elements per row, tile_y rows, global row stride stride_elems
//  - LDS padding: pad_amount_code+1 DWORDs inserted every 2^(pad_ic+1) DWORDs
// Groups 2/3 zero (2D tensor, no iteration/gather).  Issue once per block
// (wave 0); EXEC is ignored by tensor ops.
// ---------------------------------------------------------------------------
#if HAVE_TDM
__device__ __forceinline__ void tdm_load_2d(void* lds, const void* g,
                                            unsigned tile_x, unsigned tile_y,
                                            unsigned stride_elems,
                                            unsigned pad_ic, unsigned pad_ac)
{
  unsigned long long ga = (unsigned long long)(uintptr_t)g;
  u32x4 g0;
  g0[0] = 1u;                                  // count=1, user descriptor
  g0[1] = (unsigned)(uintptr_t)lds;            // lds_addr (LDS byte offset)
  g0[2] = (unsigned)ga;                        // global_addr[31:0]
  g0[3] = ((unsigned)(ga >> 32) & 0x01FFFFFFu) // global_addr[56:32]
          | (2u << 30);                        // type=2 ("image")
  i32x8 g1;
  g1[0] = (int)((1u << 16)                     // data_size = 2 bytes
                | (1u << 20)                   // pad_enable
                | (pad_ic << 22)               // pad_interval
                | (pad_ac << 25));             // pad_amount
  g1[1] = (int)((stride_elems & 0xFFFFu) << 16);          // tensor_dim0 lo (=stride)
  g1[2] = (int)((stride_elems >> 16) & 0xFFFFu);          // tensor_dim0 hi | dim1 lo(0)
  g1[3] = (int)(0x10u | (tile_x << 16));                  // dim1 hi (1M rows) | tile_dim0
  g1[4] = (int)(tile_y & 0xFFFFu);                        // tile_dim1 | tile_dim2=0
  g1[5] = (int)stride_elems;                              // tensor_dim0_stride[31:0]
  g1[6] = 0;                                              // stride hi | dim1_stride lo
  g1[7] = 0;
  i32x4 gz = {0, 0, 0, 0};
#if defined(__clang_major__) && (__clang_major__ >= 23)
  i32x8 gz8 = {0, 0, 0, 0, 0, 0, 0, 0};
  __builtin_amdgcn_tensor_load_to_lds(g0, g1, gz, gz, gz8, 0);
#else
  __builtin_amdgcn_tensor_load_to_lds(g0, g1, gz, gz, 0);
#endif
}
#endif

// Stage a [tile_y][tile_x] f16 tile into LDS (stride lds_stride halves).
// TDM path: single wave-0 DMA issue.  Fallback: per-thread async b128 copies.
__device__ __forceinline__ void stage_tile_f16(_Float16* lds, const _Float16* g,
                                               int tile_x, int tile_y,
                                               int g_stride, int lds_stride,
                                               unsigned pad_ic,
                                               int t, int nthreads, int wave)
{
#if HAVE_TDM
  (void)t; (void)nthreads;
  if (wave == 0)
    tdm_load_2d(lds, g, (unsigned)tile_x, (unsigned)tile_y,
                (unsigned)g_stride, pad_ic, 3u /* 4 DWORDs = 8 halves */);
#else
  (void)pad_ic; (void)wave;
  int cpr = tile_x / 8;                        // b128 chunks per row
  for (int s = t; s < tile_y * cpr; s += nthreads) {
    int row = s / cpr;
    int c   = s % cpr;
    async_ld_b128(lds + row * lds_stride + c * 8,
                  g + (size_t)row * g_stride + c * 8);
  }
#endif
}

__device__ __forceinline__ void stage_wait(int wave) {
#if HAVE_TDM
  if (wave == 0) wait_tensor();
#else
  (void)wave;
  wait_async();
#endif
}

__device__ __forceinline__ v16h join16(v8h lo, v8h hi) {
  return __builtin_shufflevector(lo, hi, 0,1,2,3,4,5,6,7,8,9,10,11,12,13,14,15);
}

// A fragment: 16(M) x 32(K) f16 tile from LDS, row-major base[row*ldk + k].
// Layout: lanes 0-15 row M=lane hold K 0..15; lanes 16-31 hold K 16..31 (matches frag_b).
__device__ __forceinline__ v16h frag_a(const _Float16* base, int ldk, int lane) {
  int row = lane & 15;
  int grp = (lane >> 4) << 4;               // 0 or 16
  const _Float16* r = base + row * ldk;
  v8h lo = *(const v8h*)(r + grp);          // K = grp .. grp+7
  v8h hi = *(const v8h*)(r + grp + 8);      // K = grp+8 .. grp+15
  return join16(lo, hi);
}

// B fragment: 32(K) x 16(N), tile stored transposed in LDS as [n][k] row-major.
// Lane holds column N=lane&15, 16 contiguous K starting at (lane>>4)*16.
__device__ __forceinline__ v16h frag_b(const _Float16* base, int ldk, int lane) {
  int n  = lane & 15;
  int kg = (lane >> 4) << 4;                // 0 or 16
  const _Float16* r = base + n * ldk + kg;
  v8h lo = *(const v8h*)(r);
  v8h hi = *(const v8h*)(r + 8);
  return join16(lo, hi);
}

__device__ __forceinline__ v8f wmma_f16(v16h a, v16h b, v8f c) {
  return __builtin_amdgcn_wmma_f32_16x16x32_f16(false, a, false, b, (short)0, c,
                                                false, false);
}

// ---------------------------------------------------------------------------
// Kernel 1: Q/K/V projection.  y = x @ W^T + b, output f16.
// Grid (Ed/128, B*Sd/128, 3), block 256 (8 waves). Block tile 128M x 128N,
// wave tile 32M x 64N, K staged 32 at a time via LDS (f32->f16 conversion in
// the staging path, so this stays VALU+ds).  V (z==2) is written TRANSPOSED:
// Vt[e][b*S+token], so the PV kernel can stage V tiles with the TDM.
// ---------------------------------------------------------------------------
__global__ __launch_bounds__(256) void qkv_proj_kernel(
    const float* __restrict__ X,
    const float* __restrict__ Wq, const float* __restrict__ bq,
    const float* __restrict__ Wk, const float* __restrict__ bk,
    const float* __restrict__ Wv, const float* __restrict__ bv,
    _Float16* __restrict__ Qb, _Float16* __restrict__ Kb,
    _Float16* __restrict__ Vt)
{
  const int n0 = blockIdx.x * 128;
  const int m0 = blockIdx.y * 128;
  const float* W; const float* bias; _Float16* out;
  if (blockIdx.z == 0)      { W = Wq; bias = bq; out = Qb; }
  else if (blockIdx.z == 1) { W = Wk; bias = bk; out = Kb; }
  else                      { W = Wv; bias = bv; out = Vt; }
  const bool vtrans = (blockIdx.z == 2);

  __shared__ _Float16 Xs[128 * LDK];
  __shared__ _Float16 Ws[128 * LDK];

  const int t = threadIdx.x;
  const int lane = t & 31;
  const int wave = t >> 5;
  const int wm = wave & 3;    // 4 M slots of 32 rows
  const int wn = wave >> 2;   // 2 N slots of 64 cols

  const v8f vzero = {0.f,0.f,0.f,0.f,0.f,0.f,0.f,0.f};
  v8f acc[2][4];
  for (int i = 0; i < 2; i++)
    for (int j = 0; j < 4; j++) acc[i][j] = vzero;

  for (int k0 = 0; k0 < Ed; k0 += 32) {
    // stage X tile [128 m][32 k] and W tile [128 n][32 k] as f16
    for (int s = t; s < 1024; s += 256) {
      int row = s >> 3;
      int kq  = (s & 7) * 4;
      float4 fx = *(const float4*)(X + (size_t)(m0 + row) * Ed + k0 + kq);
      v4h hx = { (_Float16)fx.x, (_Float16)fx.y, (_Float16)fx.z, (_Float16)fx.w };
      *(v4h*)(Xs + row * LDK + kq) = hx;
      float4 fw = *(const float4*)(W + (size_t)(n0 + row) * Ed + k0 + kq);
      v4h hw = { (_Float16)fw.x, (_Float16)fw.y, (_Float16)fw.z, (_Float16)fw.w };
      *(v4h*)(Ws + row * LDK + kq) = hw;
    }
    if (k0 + 32 < Ed)
      __builtin_prefetch(X + (size_t)(m0 + (t >> 1)) * Ed + k0 + 32, 0, 1);
    __syncthreads();

    v16h a[2], bf[4];
    for (int mt = 0; mt < 2; mt++)
      a[mt] = frag_a(Xs + (wm * 32 + mt * 16) * LDK, LDK, lane);
    for (int nt = 0; nt < 4; nt++)
      bf[nt] = frag_b(Ws + (wn * 64 + nt * 16) * LDK, LDK, lane);
    for (int mt = 0; mt < 2; mt++)
      for (int nt = 0; nt < 4; nt++)
        acc[mt][nt] = wmma_f16(a[mt], bf[nt], acc[mt][nt]);
    __syncthreads();
  }

  const int lrow = (lane >> 4) << 3;
  const int lcol = lane & 15;
  for (int mt = 0; mt < 2; mt++) {
    int rbase = m0 + wm * 32 + mt * 16 + lrow;
    for (int nt = 0; nt < 4; nt++) {
      int col = n0 + wn * 64 + nt * 16 + lcol;
      float bb = bias[col];
      for (int r = 0; r < 8; r++) {
        _Float16 hv = (_Float16)(acc[mt][nt][r] + bb);
        if (vtrans)
          out[(size_t)col * (Bd * Sd) + rbase + r] = hv;   // Vt[e][b*S+token]
        else
          out[(size_t)(rbase + r) * Ed + col] = hv;
      }
    }
  }
}

// ---------------------------------------------------------------------------
// Kernel 2: causal scores + softmax.  Writes P = exp(s - rowmax) (f16) and
// 1/rowsum (f32).  Grid (Sd/32, B), block 64 (2 waves, 16 q-rows each).
// Q tile staged once (TDM as 64 half-rows x 512 halves, pad every 256 DWORDs);
// K streamed in 64x64 TDM chunks.  Two passes: max, then exp/sum/store.
// ---------------------------------------------------------------------------
__global__ __launch_bounds__(64) void attn_softmax_kernel(
    const _Float16* __restrict__ Qb, const _Float16* __restrict__ Kb,
    _Float16* __restrict__ Pb, float* __restrict__ Linv)
{
  const int i0 = blockIdx.x * 32;   // query tile start within batch
  const int b  = blockIdx.y;
  const int t  = threadIdx.x;
  const int lane = t & 31;
  const int wave = t >> 5;          // 0..1

  extern __shared__ _Float16 smem[];
  _Float16* Qs = smem;              // [32 rows][QROW] = [64 half-rows][LQH]
  _Float16* Ks = smem + 32 * QROW;  // [64][LKC]

  // stage Q rows once: 64 half-rows of 512 halves, LDS stride LQH
  const _Float16* Qg = Qb + (size_t)(b * Sd + i0) * Ed;
  stage_tile_f16(Qs, Qg, 512, 64, 512, LQH, 7u, t, 64, wave);
  stage_wait(wave);
  __syncthreads();

  const int njt = (i0 >> 6) + 1;    // causal j-tile count (64-key tiles)
  const float SCALE = 0.03125f;     // 1/sqrt(1024)
  const int lrow = (lane >> 4) << 3;
  const int lcol = lane & 15;
  const v8f vzero = {0.f,0.f,0.f,0.f,0.f,0.f,0.f,0.f};

  float mrow[8], lsum[8];
  for (int r = 0; r < 8; r++) { mrow[r] = -3.0e38f; lsum[r] = 0.f; }

  for (int pass = 0; pass < 2; pass++) {
    for (int jt = 0; jt < njt; jt++) {
      v8f acc[4];
      for (int nt = 0; nt < 4; nt++) acc[nt] = vzero;

      for (int ec = 0; ec < 16; ec++) {   // E in chunks of 64
        __syncthreads();
        const _Float16* Kg = Kb + (size_t)(b * Sd + jt * 64) * Ed + ec * 64;
        stage_tile_f16(Ks, Kg, 64, 64, Ed, LKC, 4u, t, 64, wave);
        stage_wait(wave);
        __syncthreads();
        for (int ks = 0; ks < 2; ks++) {
          int e    = ec * 64 + ks * 32;
          int qoff = e + (e >= 512 ? 8 : 0);   // half-row padding skip
          v16h aq = frag_a(Qs + (wave * 16) * QROW + qoff, QROW, lane);
          for (int nt = 0; nt < 4; nt++) {
            v16h bk = frag_b(Ks + (nt * 16) * LKC + ks * 32, LKC, lane);
            acc[nt] = wmma_f16(aq, bk, acc[nt]);
          }
        }
      }

      for (int nt = 0; nt < 4; nt++) {
        int col = jt * 64 + nt * 16 + lcol;
        for (int r = 0; r < 8; r++) {
          int q = i0 + wave * 16 + lrow + r;
          float sv = acc[nt][r] * SCALE;
          bool valid = (col <= q);
          if (pass == 0) {
            if (valid) mrow[r] = fmaxf(mrow[r], sv);
          } else {
            float p = valid ? __expf(sv - mrow[r]) : 0.f;
            lsum[r] += p;
            Pb[(size_t)(b * Sd + q) * Sd + col] = (_Float16)p;
          }
        }
      }
    }
    if (pass == 0) {
      // cross-lane max within each 16-lane half (same row set)
      for (int r = 0; r < 8; r++)
        for (int off = 1; off < 16; off <<= 1)
          mrow[r] = fmaxf(mrow[r], __shfl_xor(mrow[r], off, 32));
    }
  }

  for (int r = 0; r < 8; r++)
    for (int off = 1; off < 16; off <<= 1)
      lsum[r] += __shfl_xor(lsum[r], off, 32);
  if (lcol == 0)
    for (int r = 0; r < 8; r++)
      Linv[b * Sd + i0 + wave * 16 + lrow + r] = 1.0f / lsum[r];

  // zero-fill one extra 64-key tile so the PV kernel's clipped K loop only
  // ever reads initialized probabilities
  int zc0 = njt * 64;
  if (zc0 < Sd) {
    const _Float16 hz = (_Float16)0.f;
    const v8h z = {hz,hz,hz,hz,hz,hz,hz,hz};
    for (int s = t; s < 32 * 8; s += 64) {
      int row = s >> 3;
      int c0  = (s & 7) * 8;
      *(v8h*)(Pb + (size_t)(b * Sd + i0 + row) * Sd + zc0 + c0) = z;
    }
  }
}

// ---------------------------------------------------------------------------
// Kernel 3: O = (P @ V) * (1/rowsum).  Grid (Ed/128, B*Sd/128), block 256.
// K loop clipped to the causal limit of the row tile.  Both P tiles and the
// (pre-transposed) V tiles are staged with single TDM descriptor loads.
// ---------------------------------------------------------------------------
__global__ __launch_bounds__(256) void pv_kernel(
    const _Float16* __restrict__ Pb, const _Float16* __restrict__ Vt,
    const float* __restrict__ Linv, float* __restrict__ out)
{
  const int n0   = blockIdx.x * 128;
  const int m0g  = blockIdx.y * 128;          // global row (B*S space)
  const int b    = m0g >> 11;
  const int mloc = m0g & (Sd - 1);
  const int kmax = mloc + 128;                // causal key limit for this tile

  __shared__ _Float16 Ps[128 * LDK];
  __shared__ _Float16 Vs[128 * LDK];          // Vs[e][key] (Vt already transposed)

  const int t = threadIdx.x, lane = t & 31, wave = t >> 5;
  const int wm = wave & 3, wn = wave >> 2;

  const v8f vzero = {0.f,0.f,0.f,0.f,0.f,0.f,0.f,0.f};
  v8f acc[2][4];
  for (int i = 0; i < 2; i++)
    for (int j = 0; j < 4; j++) acc[i][j] = vzero;

  const _Float16* Pg0 = Pb + (size_t)(b * Sd + mloc) * Sd;
  const _Float16* Vg0 = Vt + (size_t)n0 * (Bd * Sd) + b * Sd;

  for (int k0 = 0; k0 < kmax; k0 += 32) {
    stage_tile_f16(Ps, Pg0 + k0, 32, 128, Sd,      LDK, 3u, t, 256, wave);
    stage_tile_f16(Vs, Vg0 + k0, 32, 128, Bd * Sd, LDK, 3u, t, 256, wave);
    if (k0 + 32 < kmax)
      __builtin_prefetch(Pg0 + (size_t)(t >> 1) * Sd + k0 + 32, 0, 1);
    stage_wait(wave);
    __syncthreads();

    v16h a[2], bf[4];
    for (int mt = 0; mt < 2; mt++)
      a[mt] = frag_a(Ps + (wm * 32 + mt * 16) * LDK, LDK, lane);
    for (int nt = 0; nt < 4; nt++)
      bf[nt] = frag_b(Vs + (wn * 64 + nt * 16) * LDK, LDK, lane);
    for (int mt = 0; mt < 2; mt++)
      for (int nt = 0; nt < 4; nt++)
        acc[mt][nt] = wmma_f16(a[mt], bf[nt], acc[mt][nt]);
    __syncthreads();
  }

  const int lrow = (lane >> 4) << 3, lcol = lane & 15;
  for (int mt = 0; mt < 2; mt++) {
    int rbase = m0g + wm * 32 + mt * 16 + lrow;
    for (int nt = 0; nt < 4; nt++) {
      int col = n0 + wn * 64 + nt * 16 + lcol;
      for (int r = 0; r < 8; r++)
        out[(size_t)(rbase + r) * Ed + col] = acc[mt][nt][r] * Linv[rbase + r];
    }
  }
}

// ---------------------------------------------------------------------------
extern "C" void kernel_launch(void* const* d_in, const int* in_sizes, int n_in,
                              void* d_out, int out_size, void* d_ws, size_t ws_size,
                              hipStream_t stream) {
  (void)in_sizes; (void)n_in; (void)out_size; (void)ws_size;
  const float* X  = (const float*)d_in[0];
  // d_in[1] = causal mask — structure is hardcoded in the kernels
  const float* Wq = (const float*)d_in[2];
  const float* bq = (const float*)d_in[3];
  const float* Wk = (const float*)d_in[4];
  const float* bk = (const float*)d_in[5];
  const float* Wv = (const float*)d_in[6];
  const float* bv = (const float*)d_in[7];
  float* out = (float*)d_out;

  // workspace layout (f16 Q/K/Vt, f16 probs, f32 inv row sums): ~80 MB
  _Float16* Qb = (_Float16*)d_ws;
  _Float16* Kb = Qb + (size_t)Bd * Sd * Ed;
  _Float16* Vt = Kb + (size_t)Bd * Sd * Ed;
  _Float16* Pb = Vt + (size_t)Bd * Sd * Ed;
  float*  Linv = (float*)(Pb + (size_t)Bd * Sd * Sd);

  qkv_proj_kernel<<<dim3(Ed / 128, (Bd * Sd) / 128, 3), 256, 0, stream>>>(
      X, Wq, bq, Wk, bk, Wv, bv, Qb, Kb, Vt);

  size_t smem2 = (size_t)(32 * QROW + 64 * LKC) * sizeof(_Float16);
  attn_softmax_kernel<<<dim3(Sd / 32, Bd), 64, smem2, stream>>>(Qb, Kb, Pb, Linv);

  pv_kernel<<<dim3(Ed / 128, (Bd * Sd) / 128), 256, 0, stream>>>(Pb, Vt, Linv, out);
}
